// MultiHeadedAttention_65764539236447
// MI455X (gfx1250) — matrix operand order repaired and checked by reading, not compiled, not run on hardware
//
#include <hip/hip_runtime.h>
#include <hip/hip_bf16.h>
#include <stdint.h>

// ---------------------------------------------------------------------------
// MHA forward for MI455X (gfx1250): bf16 WMMA everywhere, f32 accumulate.
//   x:[2,2048,1024] f32, W*:[1024,1024] f32, b*:[1024] f32 -> out f32.
// Pipeline:
//   1) convert x, Wq, Wk, Wv, Wo to bf16 in workspace
//   2) fused QKV projection (one pass over xb, 24 WMMAs per k-step/wave)
//   3) flash-attention per (batch, head)    (bf16 ctx out)
//   4) out = gemm_bt(ctx, Wob) + bo         (f32 out)
// ---------------------------------------------------------------------------

typedef __bf16 bf16_t;
typedef __attribute__((ext_vector_type(16))) __bf16 v16bf;
typedef __attribute__((ext_vector_type(8)))  __bf16 v8bf;
typedef __attribute__((ext_vector_type(8)))  float  v8f;

#define SEQ   2048
#define HID   1024
#define NHEAD 16
#define DHEAD 64
#define ROWS  4096   // 2 * 2048

__device__ __forceinline__ unsigned bfbits(float f) {
  unsigned u = __float_as_uint(f);
  return (u + 0x7FFFu + ((u >> 16) & 1u)) >> 16;  // RNE fp32 -> bf16 bits
}
__device__ __forceinline__ bf16_t f2bf(float f) {
  unsigned short s = (unsigned short)bfbits(f);
  return __builtin_bit_cast(bf16_t, s);
}

// --- WMMA fragment loaders (layouts per CDNA5 ISA 7.12.2, wave32) -----------
// A 16x32 bf16: lane m = lane&15; elems 0..7 -> K = kh..kh+7, elems 8..15 ->
// K = 16+kh..16+kh+7, with kh = (lane>>4)*8. Two 16B contiguous loads.
__device__ __forceinline__ v16bf load_fragA(const bf16_t* __restrict__ base,
                                            int ld, int row0, int k0, int lane) {
  const bf16_t* p = base + (size_t)(row0 + (lane & 15)) * ld + k0 + ((lane >> 4) << 3);
  v8bf lo = *(const v8bf*)p;
  v8bf hi = *(const v8bf*)(p + 16);
  v16bf r;
#pragma unroll
  for (int i = 0; i < 8; ++i) { r[i] = lo[i]; r[8 + i] = hi[i]; }
  return r;
}

// B 32x16 bf16 loaded from a row-major "Bt"[N][K] (i.e. B = Bt^T):
// lane n = lane&15; elems 0..7 -> K = kh..kh+7, elems 8..15 -> K = 16+kh..16+kh+7,
// with kh = (lane>>4)*8. Two 16B contiguous loads.
__device__ __forceinline__ v16bf load_fragB(const bf16_t* __restrict__ baseT,
                                            int ld, int col0, int k0, int lane) {
  const bf16_t* p = baseT + (size_t)(col0 + (lane & 15)) * ld + k0 + ((lane >> 4) << 3);
  v8bf lo = *(const v8bf*)p;
  v8bf hi = *(const v8bf*)(p + 16);
  v16bf r;
#pragma unroll
  for (int i = 0; i < 8; ++i) { r[i] = lo[i]; r[8 + i] = hi[i]; }
  return r;
}

__device__ __forceinline__ v8f wmma_bf16(v16bf a, v16bf b, v8f c) {
  return __builtin_amdgcn_wmma_f32_16x16x32_bf16(false, a, false, b, (short)0, c,
                                                 false, false);
}

// --- 16-lane reductions via DPP (no LDS, no DScnt stalls) -------------------
constexpr int dpp8_sel_xor(int k) {
  int s = 0;
  for (int i = 0; i < 8; ++i) s |= ((i ^ k) & 7) << (3 * i);
  return s;
}
template <int XK>
__device__ __forceinline__ float dpp8_xor(float v) {
  return __int_as_float(
      __builtin_amdgcn_mov_dpp8(__float_as_int(v), dpp8_sel_xor(XK)));
}
__device__ __forceinline__ float dpp_row_mirror(float v) {
  return __int_as_float(__builtin_amdgcn_update_dpp(
      0, __float_as_int(v), 0x140 /*ROW_MIRROR*/, 0xf, 0xf, true));
}
__device__ __forceinline__ float halfmax16(float v) {
  v = fmaxf(v, dpp8_xor<1>(v));
  v = fmaxf(v, dpp8_xor<2>(v));
  v = fmaxf(v, dpp8_xor<4>(v));
  v = fmaxf(v, dpp_row_mirror(v));
  return v;
}
__device__ __forceinline__ float halfsum16(float v) {
  v += dpp8_xor<1>(v);
  v += dpp8_xor<2>(v);
  v += dpp8_xor<4>(v);
  v += dpp_row_mirror(v);
  return v;
}

// ---------------------------------------------------------------------------
__global__ __launch_bounds__(256)
void f32_to_bf16_kernel(const float* __restrict__ in, bf16_t* __restrict__ out, int n) {
  int i = (blockIdx.x * blockDim.x + threadIdx.x) * 4;
  if (i < n) {
    float4 f = *(const float4*)(in + i);
    uint2 p;
    p.x = bfbits(f.x) | (bfbits(f.y) << 16);
    p.y = bfbits(f.z) | (bfbits(f.w) << 16);
    *(uint2*)(out + i) = p;  // one b64 store
  }
}

// ---------------------------------------------------------------------------
// Fused QKV projection: Q/K/V[M,N] = A[M,K] * W{q,k,v}[N,K]^T + b{q,k,v}.
// Block = 128 thr (4 waves), wave tile = 32 rows x 64 cols x 3 outputs:
// per k-step 2 A-frags + 12 B-frags -> 24 WMMAs. Grid: (N/128, M/64).
// ---------------------------------------------------------------------------
__global__ __launch_bounds__(128)
void qkv_gemm_wmma(const bf16_t* __restrict__ A,
                   const bf16_t* __restrict__ Wq, const bf16_t* __restrict__ Wk,
                   const bf16_t* __restrict__ Wv,
                   const float* __restrict__ bq, const float* __restrict__ bk,
                   const float* __restrict__ bv,
                   bf16_t* __restrict__ Qo, bf16_t* __restrict__ Ko,
                   bf16_t* __restrict__ Vo) {
  const int K = HID, N = HID;
  const int lane = threadIdx.x & 31;
  const int wave = threadIdx.x >> 5;
  const int row0 = blockIdx.y * 64 + (wave >> 1) * 32;
  const int col0 = blockIdx.x * 128 + (wave & 1) * 64;

  v8f acc[3][2][4] = {};
  for (int k0 = 0; k0 < K; k0 += 32) {
    if (k0 + 64 < K) {
      __builtin_prefetch(A + (size_t)(row0 + (lane & 15)) * K + k0 + 64, 0, 0);
    }
    v16bf a0 = load_fragA(A, K, row0,      k0, lane);
    v16bf a1 = load_fragA(A, K, row0 + 16, k0, lane);
#pragma unroll
    for (int j = 0; j < 4; ++j) {
      v16bf bfq = load_fragB(Wq, K, col0 + j * 16, k0, lane);
      acc[0][0][j] = wmma_bf16(a0, bfq, acc[0][0][j]);
      acc[0][1][j] = wmma_bf16(a1, bfq, acc[0][1][j]);
      v16bf bfk = load_fragB(Wk, K, col0 + j * 16, k0, lane);
      acc[1][0][j] = wmma_bf16(a0, bfk, acc[1][0][j]);
      acc[1][1][j] = wmma_bf16(a1, bfk, acc[1][1][j]);
      v16bf bfv = load_fragB(Wv, K, col0 + j * 16, k0, lane);
      acc[2][0][j] = wmma_bf16(a0, bfv, acc[2][0][j]);
      acc[2][1][j] = wmma_bf16(a1, bfv, acc[2][1][j]);
    }
  }

  const int rb = (lane >> 4) << 3;
  const int n  = lane & 15;
#pragma unroll
  for (int m = 0; m < 3; ++m) {
    bf16_t* O = (m == 0) ? Qo : (m == 1) ? Ko : Vo;
    const float* bi = (m == 0) ? bq : (m == 1) ? bk : bv;
#pragma unroll
    for (int i = 0; i < 2; ++i) {
#pragma unroll
      for (int j = 0; j < 4; ++j) {
        float bvv = bi[col0 + j * 16 + n];
#pragma unroll
        for (int r = 0; r < 8; ++r) {
          size_t off = (size_t)(row0 + i * 16 + rb + r) * N + col0 + j * 16 + n;
          O[off] = f2bf(acc[m][i][j][r] + bvv);
        }
      }
    }
  }
}

// ---------------------------------------------------------------------------
// Generic C[M,N] = A[M,K] * Bt[N,K]^T + bias[N], f32 store (O-projection).
// Wave tile 32x64. Grid: (N/128, M/64), block 128.
// ---------------------------------------------------------------------------
__global__ __launch_bounds__(128)
void gemm_bt_wmma_f32(const bf16_t* __restrict__ A, const bf16_t* __restrict__ Bt,
                      const float* __restrict__ bias, float* __restrict__ Cout,
                      int M, int N, int K) {
  const int lane = threadIdx.x & 31;
  const int wave = threadIdx.x >> 5;
  const int row0 = blockIdx.y * 64 + (wave >> 1) * 32;
  const int col0 = blockIdx.x * 128 + (wave & 1) * 64;

  v8f acc[2][4] = {};
  for (int k0 = 0; k0 < K; k0 += 32) {
    if (k0 + 64 < K) {
      __builtin_prefetch(A + (size_t)(row0 + (lane & 15)) * K + k0 + 64, 0, 0);
    }
    v16bf a0 = load_fragA(A, K, row0,      k0, lane);
    v16bf a1 = load_fragA(A, K, row0 + 16, k0, lane);
#pragma unroll
    for (int j = 0; j < 4; ++j) {
      v16bf b = load_fragB(Bt, K, col0 + j * 16, k0, lane);
      acc[0][j] = wmma_bf16(a0, b, acc[0][j]);
      acc[1][j] = wmma_bf16(a1, b, acc[1][j]);
    }
  }

  const int rb = (lane >> 4) << 3;
  const int n  = lane & 15;
#pragma unroll
  for (int i = 0; i < 2; ++i) {
#pragma unroll
    for (int j = 0; j < 4; ++j) {
      float bv = bias[col0 + j * 16 + n];
#pragma unroll
      for (int r = 0; r < 8; ++r) {
        size_t off = (size_t)(row0 + i * 16 + rb + r) * N + col0 + j * 16 + n;
        Cout[off] = acc[i][j][r] + bv;
      }
    }
  }
}

// ---------------------------------------------------------------------------
// Flash attention. Block = 256 thr (8 waves), each wave owns 16 query rows
// (128 rows / block). Key tile = 64 keys/iter: 8 S-WMMAs + 8 O-WMMAs per
// barrier round. grid = (SEQ/128, 2*NHEAD). Q/K/V/ctx: [4096][1024] bf16
// with head h occupying columns h*64 .. h*64+63.
// ---------------------------------------------------------------------------
__global__ __launch_bounds__(256)
void attn_wmma(const bf16_t* __restrict__ Q, const bf16_t* __restrict__ Km,
               const bf16_t* __restrict__ V, bf16_t* __restrict__ ctx) {
  __shared__ bf16_t Vt[DHEAD][72];      // V tile transposed [d][key], 16B-aligned rows
  __shared__ bf16_t Ps[8][16][64];      // per-wave P tile (C-layout -> A-layout relay)

  const int lane = threadIdx.x & 31;
  const int wave = threadIdx.x >> 5;
  const int tid  = threadIdx.x;
  const int b    = blockIdx.y >> 4;
  const int h    = blockIdx.y & 15;
  const size_t base = (size_t)b * SEQ * HID + (size_t)h * DHEAD;
  const bf16_t* Qb = Q  + base;
  const bf16_t* Kb = Km + base;
  const bf16_t* Vb = V  + base;
  const int qrow = blockIdx.x * 128 + wave * 16;

  // Q fragments for this wave's 16 rows: d = 0..31 and 32..63
  v16bf qa0 = load_fragA(Qb, HID, qrow, 0,  lane);
  v16bf qa1 = load_fragA(Qb, HID, qrow, 32, lane);

  v8f o[4] = {};
  float mrow[8], lrow[8];
#pragma unroll
  for (int r = 0; r < 8; ++r) { mrow[r] = -1e30f; lrow[r] = 0.0f; }

  const float scale = 0.125f;  // 1/sqrt(64)

  for (int kt = 0; kt < SEQ; kt += 64) {
    __syncthreads();
    // stage V tile transposed: 16B global reads, 16-bit LDS transpose scatter
    for (int u = tid; u < 64 * 8; u += 256) {
      int key = u >> 3;
      int dg  = (u & 7) * 8;
      v8bf vv = *(const v8bf*)(Vb + (size_t)(kt + key) * HID + dg);
#pragma unroll
      for (int i = 0; i < 8; ++i) Vt[dg + i][key] = vv[i];
    }
    __syncthreads();

    // S tiles: 4 x 16 keys; K rows read directly from global (L2-resident)
    v8f s[4] = {};
#pragma unroll
    for (int t = 0; t < 4; ++t) {
      v16bf kb0 = load_fragB(Kb, HID, kt + t * 16, 0,  lane);
      s[t] = wmma_bf16(qa0, kb0, s[t]);
      v16bf kb1 = load_fragB(Kb, HID, kt + t * 16, 32, lane);
      s[t] = wmma_bf16(qa1, kb1, s[t]);
    }

    // online softmax per row (row r of this half-wave, column = lane&15)
    const int rb = (lane >> 4) << 3;
    const int n  = lane & 15;
#pragma unroll
    for (int r = 0; r < 8; ++r) {
      float a0 = s[0][r] * scale;
      float a1 = s[1][r] * scale;
      float a2 = s[2][r] * scale;
      float a3 = s[3][r] * scale;
      float rm = halfmax16(fmaxf(fmaxf(a0, a1), fmaxf(a2, a3)));
      float mn = fmaxf(mrow[r], rm);
      float alpha = __expf(mrow[r] - mn);
      float p0 = __expf(a0 - mn);
      float p1 = __expf(a1 - mn);
      float p2 = __expf(a2 - mn);
      float p3 = __expf(a3 - mn);
      float rs = halfsum16((p0 + p1) + (p2 + p3));
      lrow[r] = lrow[r] * alpha + rs;
      mrow[r] = mn;
#pragma unroll
      for (int j = 0; j < 4; ++j) o[j][r] *= alpha;
      Ps[wave][rb + r][n]      = f2bf(p0);
      Ps[wave][rb + r][16 + n] = f2bf(p1);
      Ps[wave][rb + r][32 + n] = f2bf(p2);
      Ps[wave][rb + r][48 + n] = f2bf(p3);
    }
    __builtin_amdgcn_wave_barrier();  // LDS ops are in-order per wave (DScnt)

    // P (16x64, A-layout) x V (64x64, via transposed LDS tile) -> O += P*V
    v16bf pa0 = load_fragA(&Ps[wave][0][0], 64, 0, 0,  lane);
    v16bf pa1 = load_fragA(&Ps[wave][0][0], 64, 0, 32, lane);
#pragma unroll
    for (int j = 0; j < 4; ++j) {
      v16bf vb0 = load_fragB(&Vt[0][0], 72, j * 16, 0,  lane);
      o[j] = wmma_bf16(pa0, vb0, o[j]);
      v16bf vb1 = load_fragB(&Vt[0][0], 72, j * 16, 32, lane);
      o[j] = wmma_bf16(pa1, vb1, o[j]);
    }
  }

  // normalize and store ctx (bf16, same [4096][1024] head-column layout)
  const int rb = (lane >> 4) << 3;
  const int n  = lane & 15;
#pragma unroll
  for (int r = 0; r < 8; ++r) {
    float linv = 1.0f / lrow[r];
    size_t off = base + (size_t)(qrow + rb + r) * HID;
#pragma unroll
    for (int j = 0; j < 4; ++j) {
      ctx[off + j * 16 + n] = f2bf(o[j][r] * linv);
    }
  }
}

// ---------------------------------------------------------------------------
extern "C" void kernel_launch(void* const* d_in, const int* in_sizes, int n_in,
                              void* d_out, int out_size, void* d_ws, size_t ws_size,
                              hipStream_t stream) {
  (void)in_sizes; (void)n_in; (void)out_size; (void)ws_size;

  const float* x  = (const float*)d_in[0];
  const float* Wq = (const float*)d_in[1];
  const float* bq = (const float*)d_in[2];
  const float* Wk = (const float*)d_in[3];
  const float* bk = (const float*)d_in[4];
  const float* Wv = (const float*)d_in[5];
  const float* bv = (const float*)d_in[6];
  const float* Wo = (const float*)d_in[7];
  const float* bo = (const float*)d_in[8];
  float* out = (float*)d_out;

  // workspace layout (bytes)
  char* ws = (char*)d_ws;
  bf16_t* xb   = (bf16_t*)(ws + 0);               //  8 MiB: 4096x1024
  bf16_t* Wqb  = (bf16_t*)(ws + (8u  << 20));     //  2 MiB each
  bf16_t* Wkb  = (bf16_t*)(ws + (10u << 20));
  bf16_t* Wvb  = (bf16_t*)(ws + (12u << 20));
  bf16_t* Wob  = (bf16_t*)(ws + (14u << 20));
  bf16_t* Qw   = (bf16_t*)(ws + (16u << 20));     //  8 MiB each
  bf16_t* Kw   = (bf16_t*)(ws + (24u << 20));
  bf16_t* Vw   = (bf16_t*)(ws + (32u << 20));
  bf16_t* ctxb = (bf16_t*)(ws + (40u << 20));     // end at 48 MiB

  const int NX = ROWS * HID;   // 4194304
  const int NW = HID * HID;    // 1048576

  // 1) fp32 -> bf16 conversions
  f32_to_bf16_kernel<<<NX / 4 / 256, 256, 0, stream>>>(x,  xb,  NX);
  f32_to_bf16_kernel<<<NW / 4 / 256, 256, 0, stream>>>(Wq, Wqb, NW);
  f32_to_bf16_kernel<<<NW / 4 / 256, 256, 0, stream>>>(Wk, Wkb, NW);
  f32_to_bf16_kernel<<<NW / 4 / 256, 256, 0, stream>>>(Wv, Wvb, NW);
  f32_to_bf16_kernel<<<NW / 4 / 256, 256, 0, stream>>>(Wo, Wob, NW);

  // 2) fused QKV projection: one pass over xb
  dim3 gg(HID / 128, ROWS / 64);  // (8, 64)
  qkv_gemm_wmma<<<gg, 128, 0, stream>>>(xb, Wqb, Wkb, Wvb, bq, bk, bv, Qw, Kw, Vw);

  // 3) attention per (batch, head): grid (SEQ/128, 2*16)
  attn_wmma<<<dim3(SEQ / 128, 2 * NHEAD), 256, 0, stream>>>(Qw, Kw, Vw, ctxb);

  // 4) output projection -> f32 d_out
  gemm_bt_wmma_f32<<<gg, 128, 0, stream>>>(ctxb, Wob, bo, out, ROWS, HID, HID);
}
